// STGODEModel_19275813224640
// MI455X (gfx1250) — matrix operand order repaired and checked by reading, not compiled, not run on hardware
//
#include <hip/hip_runtime.h>
#include <math.h>

// ---------------- problem constants (from reference) ----------------
#define BB      8
#define SS      12
#define NN      10000
#define FIN     3
#define HH      64
#define EE      640000
#define BN      (BB * NN)          // 80000 rows
#define DT_F    0.05f
#define NSTEPS  20

typedef _Float16 f16;
typedef __attribute__((ext_vector_type(16))) _Float16 v16h;
typedef __attribute__((ext_vector_type(8)))  float    v8f;

enum {
    MODE_PLAIN = 0, MODE_RELU = 1, MODE_TANH = 2,
    MODE_GCN = 3, MODE_GATE_STEP = 4, MODE_GATE_FINAL = 5
};

// =====================================================================
// WMMA GEMM: out[80000 x 64] = epilogue( A[80000 x (64*KC/2)] @ W + b )
// 8 waves / block, each wave owns a 16-row strip (block = 128 rows).
// A is staged (f32 -> f16) through LDS with coalesced loads, then read
// back per-lane in the v_wmma 16-bit A-fragment swizzle.
// W is pre-packed (pack_w) into per-lane B-fragment order: one 32B
// contiguous chunk per lane per (ntile, kchunk) -> two b128 loads.
// =====================================================================
__global__ __launch_bounds__(256)
void gemm_h64(const float* __restrict__ A1, const float* __restrict__ A2,
              const f16* __restrict__ Wpk, const float* __restrict__ bias,
              int actRelu, int kcTotal, int mode,
              const float* __restrict__ selfnorm,
              float* __restrict__ out0, float* __restrict__ out1,
              const float* __restrict__ P0, float* __restrict__ Q0,
              float* __restrict__ Q1,
              float c0, float cacc, int accInit)
{
    __shared__ f16 Alds[8][16][64];
    const int tid = threadIdx.x;
    const int w   = tid >> 5;
    const int l   = tid & 31;
    const int rowbase = blockIdx.x * 128 + w * 16;

    v8f acc[4] = {};                      // 4 N-tiles of 16x16 f32 accum
    const int m     = l & 15;             // A row within strip
    const int koff8 = (l >> 4) * 8;       // upper half-wave K offset

    const int nSeg = kcTotal >> 1;        // 1 (K=64) or 2 (K=128)
    for (int seg = 0; seg < nSeg; ++seg) {
        const float* __restrict__ A = (seg == 0) ? A1 : A2;
        __syncthreads();                  // protect LDS reuse across segs
        // ---- stage 16 rows x 64 cols, coalesced float2, cvt to f16 ----
        #pragma unroll
        for (int r = 0; r < 16; ++r) {
            float2 v = *(const float2*)&A[(size_t)(rowbase + r) * HH + 2 * l];
            float x0 = v.x, x1 = v.y;
            if (actRelu) { x0 = fmaxf(x0, 0.f); x1 = fmaxf(x1, 0.f); }
            union { f16 h[2]; unsigned u; } p;
            p.h[0] = (f16)x0; p.h[1] = (f16)x1;
            *(unsigned*)&Alds[w][r][2 * l] = p.u;
        }
        __syncthreads();
        // ---- two K=32 chunks per segment ----
        #pragma unroll
        for (int kc2 = 0; kc2 < 2; ++kc2) {
            const int kc = seg * 2 + kc2;
            union { v16h v; unsigned u[8]; } af;
            #pragma unroll
            for (int j = 0; j < 8; ++j) {
                int k = kc2 * 32 + ((j & 4) ? 16 : 0) + koff8 + 2 * (j & 3);
                af.u[j] = *(const unsigned*)&Alds[w][m][k];
            }
            #pragma unroll
            for (int nt = 0; nt < 4; ++nt) {
                union { v16h v; uint4 q[2]; } bf;
                const f16* wp = &Wpk[(((size_t)nt * kcTotal + kc) * 32 + l) * 16];
                bf.q[0] = *(const uint4*)wp;
                bf.q[1] = *(const uint4*)(wp + 8);
                acc[nt] = __builtin_amdgcn_wmma_f32_16x16x32_f16(
                    false, af.v, false, bf.v, (short)0, acc[nt], false, false);
            }
        }
    }

    // ---- epilogue: C layout VGPR r -> row = base + r + (lane>=16)*8 ----
    #pragma unroll
    for (int nt = 0; nt < 4; ++nt) {
        #pragma unroll
        for (int r = 0; r < 8; ++r) {
            const int row = rowbase + r + ((l >> 4) << 3);
            const int col = nt * 16 + (l & 15);
            const size_t idx = (size_t)row * HH + col;
            float v = acc[nt][r];
            switch (mode) {
            case MODE_PLAIN: out0[idx] = v + bias[col]; break;
            case MODE_RELU:  out0[idx] = fmaxf(v + bias[col], 0.f); break;
            case MODE_TANH:  out0[idx] = tanhf(v + bias[col]); break;
            case MODE_GCN:   // keep raw xw for edge scatter + fused self-loop term
                out0[idx] = v;
                out1[idx] = v * selfnorm[row] + bias[col];
                break;
            case MODE_GATE_STEP: {   // fused sigmoid gate + RK4 axpy
                float s = 1.f / (1.f + __expf(-(v + bias[col])));
                float fo = s * A1[idx] + (1.f - s) * A2[idx];   // A1=g, A2=m
                Q0[idx] = P0[idx] + c0 * fo;                    // ht = h + c0*k
                float a = accInit ? 0.f : Q1[idx];
                Q1[idx] = a + cacc * fo;                        // acc += w*k
            } break;
            case MODE_GATE_FINAL: {  // h = h + DT/6 * (acc + k4)
                float s = 1.f / (1.f + __expf(-(v + bias[col])));
                float fo = s * A1[idx] + (1.f - s) * A2[idx];
                Q0[idx] = P0[idx] + c0 * (Q1[idx] + fo);
            } break;
            }
        }
    }
}

// Pack f32 W[K x 64] -> f16 B-fragments [nt][kc][lane][16]
__global__ void pack_w(const float* __restrict__ W, f16* __restrict__ dst, int KC)
{
    int t = blockIdx.x * blockDim.x + threadIdx.x;
    if (t >= 2048 * KC) return;
    int e    = t & 15;
    int lane = (t >> 4) & 31;
    int kc   = (t >> 9) % KC;
    int nt   = t / (512 * KC);
    int k = kc * 32 + ((e & 8) ? 16 : 0) + ((lane >= 16) ? 8 : 0) + (e & 7);   // B: lanes 0-15 K=0-7,16-23; 16-31 K=8-15,24-31
    int n = nt * 16 + (lane & 15);
    dst[t] = (f16)W[k * HH + n];
}

// ---------------- GCN edge scatter: out[col] += coef * xw[row] ----------------
__global__ __launch_bounds__(256)
void gcn_scatter(const float* __restrict__ xw, float* __restrict__ out,
                 const int* __restrict__ row, const int* __restrict__ col,
                 const float* __restrict__ coef)
{
    int t = blockIdx.x * blockDim.x + threadIdx.x;
    int e = t >> 6, f = t & 63;
    if (e >= EE) return;
    float v = coef[e] * xw[(size_t)row[e] * HH + f];
    atomicAdd(&out[(size_t)col[e] * HH + f], v);
}

// ---------------- small setup kernels ----------------
__global__ void fill_ones(float* p, int n)
{ int i = blockIdx.x * blockDim.x + threadIdx.x; if (i < n) p[i] = 1.f; }

__global__ void deg_scatter(float* deg, const int* col, const float* ew)
{ int i = blockIdx.x * blockDim.x + threadIdx.x; if (i < EE) atomicAdd(&deg[col[i]], ew[i]); }

__global__ void coef_k(float* coef, const float* deg, const int* row,
                       const int* col, const float* ew)
{
    int i = blockIdx.x * blockDim.x + threadIdx.x;
    if (i < EE) coef[i] = rsqrtf(deg[row[i]]) * ew[i] * rsqrtf(deg[col[i]]);
}

__global__ void inv_k(float* p, int n)     // selfnorm = dis*dis = 1/deg (in place)
{ int i = blockIdx.x * blockDim.x + threadIdx.x; if (i < n) p[i] = 1.f / p[i]; }

__global__ void relu_copy(const float* __restrict__ in, float* __restrict__ out, int n)
{ int i = blockIdx.x * blockDim.x + threadIdx.x; if (i < n) out[i] = fmaxf(in[i], 0.f); }

// encoder layer 1 (K=3, trivial VALU)
__global__ void enc1_k(const float* __restrict__ X, const float* __restrict__ W1,
                       const float* __restrict__ b1, float* __restrict__ out)
{
    int t = blockIdx.x * blockDim.x + threadIdx.x;
    if (t >= BN * HH) return;
    int i = t >> 6, j = t & 63;
    int b = i / NN, n = i % NN;
    const float* xp = &X[(((size_t)b * SS + (SS - 1)) * NN + n) * FIN];
    float v = b1[j] + xp[0] * W1[0 * HH + j] + xp[1] * W1[1 * HH + j] + xp[2] * W1[2 * HH + j];
    out[(size_t)i * HH + j] = fmaxf(v, 0.f);
}

// decoder: out[i] = dot(t[i,:], w2) + b2   (wave-per-node reduction)
__global__ __launch_bounds__(256)
void dec_reduce(const float* __restrict__ t, const float* __restrict__ w2,
                const float* __restrict__ b2, float* __restrict__ out)
{
    int w = threadIdx.x >> 5, l = threadIdx.x & 31;
    int node = blockIdx.x * 8 + w;
    if (node >= BN) return;
    float s = t[(size_t)node * HH + l] * w2[l] + t[(size_t)node * HH + 32 + l] * w2[32 + l];
    #pragma unroll
    for (int off = 16; off; off >>= 1) s += __shfl_xor(s, off, 32);
    if (l == 0) out[node] = s + b2[0];
}

// =====================================================================
extern "C" void kernel_launch(void* const* d_in, const int* in_sizes, int n_in,
                              void* d_out, int out_size, void* d_ws, size_t ws_size,
                              hipStream_t stream)
{
    (void)in_sizes; (void)n_in; (void)out_size; (void)ws_size;
    const float* X       = (const float*)d_in[0];
    const float* ew      = (const float*)d_in[1];
    const float* enc_w1  = (const float*)d_in[2];
    const float* enc_b1  = (const float*)d_in[3];
    const float* enc_w2  = (const float*)d_in[4];
    const float* enc_b2  = (const float*)d_in[5];
    const float* gcn_w   = (const float*)d_in[6];
    const float* gcn_b   = (const float*)d_in[7];
    const float* gc1_w   = (const float*)d_in[8];
    const float* gc1_b   = (const float*)d_in[9];
    const float* gc2_w   = (const float*)d_in[10];
    const float* gc2_b   = (const float*)d_in[11];
    const float* mlp_w1  = (const float*)d_in[12];
    const float* mlp_b1  = (const float*)d_in[13];
    const float* mlp_w2  = (const float*)d_in[14];
    const float* mlp_b2  = (const float*)d_in[15];
    const float* gate_w  = (const float*)d_in[16];
    const float* gate_b  = (const float*)d_in[17];
    const float* dec_w1  = (const float*)d_in[18];
    const float* dec_b1  = (const float*)d_in[19];
    const float* dec_w2  = (const float*)d_in[20];
    const float* dec_b2  = (const float*)d_in[21];
    const int*   ei      = (const int*)d_in[22];
    const int* row = ei;
    const int* col = ei + EE;
    float* out = (float*)d_out;

    // ---- workspace carve-up ----
    float* f = (float*)d_ws;
    float* h    = f; f += (size_t)BN * HH;
    float* ht   = f; f += (size_t)BN * HH;
    float* racc = f; f += (size_t)BN * HH;
    float* t0   = f; f += (size_t)BN * HH;
    float* g1   = f; f += (size_t)BN * HH;
    float* g    = f; f += (size_t)BN * HH;
    float* mb   = f; f += (size_t)BN * HH;
    float* selfnorm = f; f += BN;     // deg -> 1/deg
    float* coef = f; f += EE;
    f16* pk = (f16*)f;
    f16* enc2pk = pk; pk += 4096;
    f16* gcnpk  = pk; pk += 4096;
    f16* gc1pk  = pk; pk += 4096;
    f16* gc2pk  = pk; pk += 4096;
    f16* mlp1pk = pk; pk += 4096;
    f16* mlp2pk = pk; pk += 4096;
    f16* dec1pk = pk; pk += 4096;
    f16* gatepk = pk; pk += 8192;

    const int GEMM_GRID = BN / 128;               // 625
    const int EW_GRID   = (EE * HH) / 256;        // edge*feat threads
    const int EL_GRID   = (BN * HH) / 256;

    auto gemm = [&](const float* A1, const float* A2, const f16* W, const float* b,
                    int actRelu, int kcTotal, int mode, float* o0, float* o1,
                    const float* P0, float* Q0, float* Q1,
                    float c0, float cacc, int accInit) {
        hipLaunchKernelGGL(gemm_h64, dim3(GEMM_GRID), dim3(256), 0, stream,
                           A1, A2, W, b, actRelu, kcTotal, mode, selfnorm,
                           o0, o1, P0, Q0, Q1, c0, cacc, accInit);
    };

    // ---- one-time prep: pack weights, degree/coef/selfnorm ----
    hipLaunchKernelGGL(pack_w, dim3(16), dim3(256), 0, stream, enc_w2, enc2pk, 2);
    hipLaunchKernelGGL(pack_w, dim3(16), dim3(256), 0, stream, gcn_w,  gcnpk,  2);
    hipLaunchKernelGGL(pack_w, dim3(16), dim3(256), 0, stream, gc1_w,  gc1pk,  2);
    hipLaunchKernelGGL(pack_w, dim3(16), dim3(256), 0, stream, gc2_w,  gc2pk,  2);
    hipLaunchKernelGGL(pack_w, dim3(16), dim3(256), 0, stream, mlp_w1, mlp1pk, 2);
    hipLaunchKernelGGL(pack_w, dim3(16), dim3(256), 0, stream, mlp_w2, mlp2pk, 2);
    hipLaunchKernelGGL(pack_w, dim3(16), dim3(256), 0, stream, dec_w1, dec1pk, 2);
    hipLaunchKernelGGL(pack_w, dim3(32), dim3(256), 0, stream, gate_w, gatepk, 4);

    hipLaunchKernelGGL(fill_ones,  dim3((BN + 255) / 256), dim3(256), 0, stream, selfnorm, BN);
    hipLaunchKernelGGL(deg_scatter, dim3((EE + 255) / 256), dim3(256), 0, stream, selfnorm, col, ew);
    hipLaunchKernelGGL(coef_k,     dim3((EE + 255) / 256), dim3(256), 0, stream, coef, selfnorm, row, col, ew);
    hipLaunchKernelGGL(inv_k,      dim3((BN + 255) / 256), dim3(256), 0, stream, selfnorm, BN);

    // ---- encoder + initial GCN ----
    hipLaunchKernelGGL(enc1_k, dim3(EL_GRID), dim3(256), 0, stream, X, enc_w1, enc_b1, mb);
    gemm(mb, nullptr, enc2pk, enc_b2, 0, 2, MODE_PLAIN, h, nullptr, nullptr, nullptr, nullptr, 0.f, 0.f, 0);
    gemm(h, nullptr, gcnpk, gcn_b, 0, 2, MODE_GCN, t0, g1, nullptr, nullptr, nullptr, 0.f, 0.f, 0);
    hipLaunchKernelGGL(gcn_scatter, dim3(EW_GRID), dim3(256), 0, stream, t0, g1, row, col, coef);
    hipLaunchKernelGGL(relu_copy,   dim3(EL_GRID), dim3(256), 0, stream, g1, h, BN * HH);

    // ---- f_ode: gate GEMM epilogue fuses sigmoid-gate + RK4 updates ----
    auto f_ode = [&](const float* hin, float c0, float cacc, int accInit, int isFinal) {
        gemm(hin, nullptr, gc1pk, gc1_b, 0, 2, MODE_GCN, t0, g1, nullptr, nullptr, nullptr, 0.f, 0.f, 0);
        hipLaunchKernelGGL(gcn_scatter, dim3(EW_GRID), dim3(256), 0, stream, t0, g1, row, col, coef);
        gemm(g1, nullptr, gc2pk, gc2_b, 1 /*relu on A*/, 2, MODE_GCN, t0, g, nullptr, nullptr, nullptr, 0.f, 0.f, 0);
        hipLaunchKernelGGL(gcn_scatter, dim3(EW_GRID), dim3(256), 0, stream, t0, g, row, col, coef);
        gemm(hin, nullptr, mlp1pk, mlp_b1, 0, 2, MODE_TANH, t0, nullptr, nullptr, nullptr, nullptr, 0.f, 0.f, 0);
        gemm(t0, nullptr, mlp2pk, mlp_b2, 0, 2, MODE_PLAIN, mb, nullptr, nullptr, nullptr, nullptr, 0.f, 0.f, 0);
        gemm(g, mb, gatepk, gate_b, 0, 4, isFinal ? MODE_GATE_FINAL : MODE_GATE_STEP,
             nullptr, nullptr, h, isFinal ? h : ht, racc, c0, cacc, accInit);
    };

    for (int s = 0; s < NSTEPS; ++s) {
        f_ode(h,  0.5f * DT_F, 1.f, 1, 0);   // k1: ht = h + DT/2*k1, acc = k1
        f_ode(ht, 0.5f * DT_F, 2.f, 0, 0);   // k2: ht = h + DT/2*k2, acc += 2k2
        f_ode(ht, DT_F,        2.f, 0, 0);   // k3: ht = h + DT*k3,   acc += 2k3
        f_ode(ht, DT_F / 6.f,  0.f, 0, 1);   // k4: h  = h + DT/6*(acc + k4)
    }

    // ---- decoder ----
    gemm(h, nullptr, dec1pk, dec_b1, 0, 2, MODE_RELU, t0, nullptr, nullptr, nullptr, nullptr, 0.f, 0.f, 0);
    hipLaunchKernelGGL(dec_reduce, dim3(BN / 8), dim3(256), 0, stream, t0, dec_w2, dec_b2, out);
}
